// MultiHeadSelfAttention_47614007443670
// MI455X (gfx1250) — matrix operand order repaired and checked by reading, not compiled, not run on hardware
//
#include <hip/hip_runtime.h>
#include <hip/hip_bf16.h>

// ---------------- types ----------------
typedef __attribute__((ext_vector_type(16))) __bf16      bf16x16;
typedef __attribute__((ext_vector_type(8)))  float       f32x8;
typedef __attribute__((ext_vector_type(8)))  unsigned short u16x8;

struct Frag2 { u16x8 lo, hi; };

static __device__ __forceinline__ bf16x16 ld_frag(const unsigned short* plo,
                                                  const unsigned short* phi) {
    Frag2 f;
    f.lo = *(const u16x8*)plo;
    f.hi = *(const u16x8*)phi;
    return __builtin_bit_cast(bf16x16, f);
}

static __device__ __forceinline__ unsigned short f32_to_bf16(float f) {
    unsigned int u = __float_as_uint(f);
    u += 0x7FFFu + ((u >> 16) & 1u);      // round-to-nearest-even
    return (unsigned short)(u >> 16);
}
static __device__ __forceinline__ float bf16_to_f32(unsigned short h) {
    return __uint_as_float(((unsigned int)h) << 16);
}

static __device__ __forceinline__ f32x8 wmma_bf16(bf16x16 a, bf16x16 b, f32x8 c) {
    return __builtin_amdgcn_wmma_f32_16x16x32_bf16(false, a, false, b,
                                                   (short)0, c, false, false);
}

// Direct global -> LDS async copy (16B per lane), ASYNCcnt-tracked.
// LDS byte offset = low 32 bits of the generic LDS address (LDS aperture rule).
static __device__ __forceinline__ void async_g2l_b128(const unsigned short* gaddr,
                                                      unsigned short* laddr) {
    unsigned lds_off = (unsigned)(unsigned long long)laddr;
    asm volatile("global_load_async_to_lds_b128 %0, %1, off"
                 :: "v"(lds_off), "v"(gaddr) : "memory");
}
static __device__ __forceinline__ void wait_async0() {
    asm volatile("s_wait_asynccnt 0x0" ::: "memory");
}

// ---------------- problem constants ----------------
#define BB 2
#define SS 2048
#define DD 1024
#define HH 16
#define DK 64

// ---------------- elementwise f32 -> bf16 ----------------
__global__ void cvt_bf16_kernel(const float* __restrict__ in,
                                unsigned short* __restrict__ out, int n) {
    int i = blockIdx.x * blockDim.x + threadIdx.x;
    if (i < n) out[i] = f32_to_bf16(in[i]);
}

// ---------------- RoPE on bf16 [BH, S, 64] ----------------
__global__ void rope_kernel(unsigned short* __restrict__ buf,
                            const int* __restrict__ pos, int total_rows) {
    int t = blockIdx.x * blockDim.x + threadIdx.x;
    int j = t & 31;              // pair index 0..31
    int row = t >> 5;            // (b*H + h)*S + s
    if (row >= total_rows) return;
    int s = row & (SS - 1);
    float p = (float)pos[s];
    float ex = -((float)(2 * j) * (1.0f / 64.0f)) * 9.2103403720f; // ln(1e4)
    float freq = __expf(ex);
    float ang = p * freq;
    float sn, cs;
    __sincosf(ang, &sn, &cs);
    size_t base = (size_t)row * DK + 2 * j;
    float x1 = bf16_to_f32(buf[base + 0]);
    float x2 = bf16_to_f32(buf[base + 1]);
    buf[base + 0] = f32_to_bf16(x1 * cs - x2 * sn);
    buf[base + 1] = f32_to_bf16(x1 * sn + x2 * cs);
}

// ---------------- bf16 GEMM:  C[m,n] = sum_k A[m,k]*Bw[n,k] ----------------
// mode 0: scatter bf16 into q/k/v [B,H,S,64]   mode 1: store f32 to outf[m*N+n]
__global__ __launch_bounds__(128) void gemm_bf16_kernel(
    const unsigned short* __restrict__ A, const unsigned short* __restrict__ Bw,
    int N, int K, int mode,
    unsigned short* __restrict__ qb, unsigned short* __restrict__ kb,
    unsigned short* __restrict__ vb, float* __restrict__ outf) {

    __shared__ __align__(16) unsigned short As[64 * 64];
    __shared__ __align__(16) unsigned short Bs[64 * 64];

    const int tid = threadIdx.x;
    const int w = tid >> 5, l = tid & 31;
    const int lm = l & 15, lh = l >> 4;
    const int m0 = blockIdx.y * 64;
    const int n0 = blockIdx.x * 64;
    const int mrow = w * 16 + lm;

    f32x8 acc0 = {}, acc1 = {}, acc2 = {}, acc3 = {};

    for (int k0 = 0; k0 < K; k0 += 64) {
#pragma unroll
        for (int i = 0; i < 4; ++i) {
            int seg = tid + i * 128;
            int r = seg >> 3, c = seg & 7;
            async_g2l_b128(&A[(size_t)(m0 + r) * K + k0 + c * 8], &As[r * 64 + c * 8]);
            async_g2l_b128(&Bw[(size_t)(n0 + r) * K + k0 + c * 8], &Bs[r * 64 + c * 8]);
        }
        wait_async0();
        __syncthreads();
#pragma unroll
        for (int ko = 0; ko < 64; ko += 32) {
            // hoist all fragments of this k-stage, then run the WMMA chain
            bf16x16 a = ld_frag(&As[mrow * 64 + ko + 8 * lh],
                                &As[mrow * 64 + ko + 16 + 8 * lh]);
            bf16x16 bf[4];
#pragma unroll
            for (int t = 0; t < 4; ++t)
                bf[t] = ld_frag(&Bs[(t * 16 + lm) * 64 + ko + 8 * lh],
                                &Bs[(t * 16 + lm) * 64 + ko + 16 + 8 * lh]);
            acc0 = wmma_bf16(a, bf[0], acc0);
            acc1 = wmma_bf16(a, bf[1], acc1);
            acc2 = wmma_bf16(a, bf[2], acc2);
            acc3 = wmma_bf16(a, bf[3], acc3);
        }
        __syncthreads();
    }

    f32x8 accs[4] = {acc0, acc1, acc2, acc3};
    if (mode == 0) {
#pragma unroll
        for (int t = 0; t < 4; ++t) {
#pragma unroll
            for (int v = 0; v < 8; ++v) {
                int m = m0 + w * 16 + v + 8 * lh;
                int n = n0 + t * 16 + lm;
                int b = m >> 11, s = m & (SS - 1);
                int tt = n >> 10, rem = n & (DD - 1);
                int hh = rem >> 6, dd = rem & 63;
                unsigned short* base = (tt == 0) ? qb : ((tt == 1) ? kb : vb);
                base[(((size_t)(b * HH + hh) * SS) + s) * DK + dd] =
                    f32_to_bf16(accs[t][v]);
            }
        }
    } else {
#pragma unroll
        for (int t = 0; t < 4; ++t) {
#pragma unroll
            for (int v = 0; v < 8; ++v) {
                int m = m0 + w * 16 + v + 8 * lh;
                int n = n0 + t * 16 + lm;
                outf[(size_t)m * N + n] = accs[t][v];
            }
        }
    }
}

// ---------------- flash attention: per (b,h,64-query tile) ----------------
__global__ __launch_bounds__(128) void fa_kernel(
    const unsigned short* __restrict__ qb, const unsigned short* __restrict__ kb,
    const unsigned short* __restrict__ vb, unsigned short* __restrict__ aob) {

    const int b = blockIdx.z, h = blockIdx.y;
    const int q0 = blockIdx.x * 64;
    const int tid = threadIdx.x;
    const int w = tid >> 5, l = tid & 31;
    const int lm = l & 15, lh = l >> 4;

    const size_t head = (size_t)(b * HH + h) * SS;
    const unsigned short* Qg = qb + head * DK;
    const unsigned short* Kg = kb + head * DK;
    const unsigned short* Vg = vb + head * DK;

    __shared__ __align__(16) unsigned short Ks[64 * 64];
    __shared__ __align__(16) unsigned short Vt[64 * 64];
    __shared__ __align__(16) unsigned short Ps[4][16 * 64];

    // stage Q through Ks via async copy, build per-wave A fragments (16 x 64)
#pragma unroll
    for (int i = 0; i < 4; ++i) {
        int seg = tid + i * 128;
        int r = seg >> 3, c = seg & 7;
        async_g2l_b128(&Qg[(size_t)(q0 + r) * DK + c * 8], &Ks[r * 64 + c * 8]);
    }
    wait_async0();
    __syncthreads();
    const int mrow = w * 16 + lm;
    bf16x16 qa0 = ld_frag(&Ks[mrow * 64 + 0 + 8 * lh], &Ks[mrow * 64 + 16 + 8 * lh]);
    bf16x16 qa1 = ld_frag(&Ks[mrow * 64 + 32 + 8 * lh], &Ks[mrow * 64 + 48 + 8 * lh]);
    __syncthreads();

    f32x8 oa0 = {}, oa1 = {}, oa2 = {}, oa3 = {};
    float mrun[8], lrun[8];
#pragma unroll
    for (int v = 0; v < 8; ++v) { mrun[v] = -3.0e38f; lrun[v] = 0.0f; }

    const int ktmax = q0 / 64;
    for (int kt = 0; kt <= ktmax; ++kt) {
        const int k0 = kt * 64;
        // K tile row-major via async; V tile transposed (row = d, col = key)
#pragma unroll
        for (int i = 0; i < 4; ++i) {
            int seg = tid + i * 128;
            int r = seg >> 3, c = seg & 7;
            async_g2l_b128(&Kg[(size_t)(k0 + r) * DK + c * 8], &Ks[r * 64 + c * 8]);
            uint4 vv = *(const uint4*)&Vg[(size_t)(k0 + r) * DK + c * 8];
            const unsigned short* e = (const unsigned short*)&vv;
#pragma unroll
            for (int j = 0; j < 8; ++j) Vt[(c * 8 + j) * 64 + r] = e[j];
        }
        wait_async0();
        __syncthreads();

        // scores: S(16x64) = Q(16x64) x K^T  -- hoist 8 B frags, then 8 WMMAs
        bf16x16 kf0[4], kf1[4];
#pragma unroll
        for (int t = 0; t < 4; ++t) {
            int nloc = t * 16 + lm;
            kf0[t] = ld_frag(&Ks[nloc * 64 + 0 + 8 * lh], &Ks[nloc * 64 + 16 + 8 * lh]);
            kf1[t] = ld_frag(&Ks[nloc * 64 + 32 + 8 * lh], &Ks[nloc * 64 + 48 + 8 * lh]);
        }
        f32x8 st[4];
#pragma unroll
        for (int t = 0; t < 4; ++t) {
            f32x8 acc = {};
            acc = wmma_bf16(qa0, kf0[t], acc);
            acc = wmma_bf16(qa1, kf1[t], acc);
            st[t] = acc;
        }
        // scale + causal mask
        const bool diag = (kt == ktmax);
#pragma unroll
        for (int t = 0; t < 4; ++t) {
#pragma unroll
            for (int v = 0; v < 8; ++v) {
                float sv = st[t][v] * 0.125f; // 1/sqrt(64)
                if (diag) {
                    int kj = k0 + t * 16 + lm;
                    int qi = q0 + w * 16 + v + 8 * lh;
                    if (kj > qi) sv = -1.0e30f;
                }
                st[t][v] = sv;
            }
        }
        // online softmax: rows live across the 16 lanes of each half-wave
        float mnew[8], alpha[8];
#pragma unroll
        for (int v = 0; v < 8; ++v) {
            float mx = fmaxf(fmaxf(st[0][v], st[1][v]), fmaxf(st[2][v], st[3][v]));
#pragma unroll
            for (int off = 1; off < 16; off <<= 1)
                mx = fmaxf(mx, __shfl_xor(mx, off, 32));
            mnew[v] = fmaxf(mrun[v], mx);
            alpha[v] = __expf(mrun[v] - mnew[v]);
        }
#pragma unroll
        for (int t = 0; t < 4; ++t)
#pragma unroll
            for (int v = 0; v < 8; ++v)
                st[t][v] = __expf(st[t][v] - mnew[v]);
#pragma unroll
        for (int v = 0; v < 8; ++v) {
            float s = st[0][v] + st[1][v] + st[2][v] + st[3][v];
#pragma unroll
            for (int off = 1; off < 16; off <<= 1)
                s += __shfl_xor(s, off, 32);
            lrun[v] = lrun[v] * alpha[v] + s;
            mrun[v] = mnew[v];
        }
#pragma unroll
        for (int v = 0; v < 8; ++v) {
            oa0[v] *= alpha[v]; oa1[v] *= alpha[v];
            oa2[v] *= alpha[v]; oa3[v] *= alpha[v];
        }
        // P: C-layout -> A-layout via per-wave LDS staging
#pragma unroll
        for (int t = 0; t < 4; ++t)
#pragma unroll
            for (int v = 0; v < 8; ++v)
                Ps[w][(v + 8 * lh) * 64 + t * 16 + lm] = f32_to_bf16(st[t][v]);
        bf16x16 pa0 = ld_frag(&Ps[w][lm * 64 + 0 + 8 * lh], &Ps[w][lm * 64 + 16 + 8 * lh]);
        bf16x16 pa1 = ld_frag(&Ps[w][lm * 64 + 32 + 8 * lh], &Ps[w][lm * 64 + 48 + 8 * lh]);
        // O += P(16x64) x V(64x64): hoist 8 B frags from transposed V tile
        bf16x16 vf0[4], vf1[4];
#pragma unroll
        for (int t = 0; t < 4; ++t) {
            int dloc = t * 16 + lm;
            vf0[t] = ld_frag(&Vt[dloc * 64 + 0 + 8 * lh], &Vt[dloc * 64 + 16 + 8 * lh]);
            vf1[t] = ld_frag(&Vt[dloc * 64 + 32 + 8 * lh], &Vt[dloc * 64 + 48 + 8 * lh]);
        }
        oa0 = wmma_bf16(pa0, vf0[0], oa0);
        oa0 = wmma_bf16(pa1, vf1[0], oa0);
        oa1 = wmma_bf16(pa0, vf0[1], oa1);
        oa1 = wmma_bf16(pa1, vf1[1], oa1);
        oa2 = wmma_bf16(pa0, vf0[2], oa2);
        oa2 = wmma_bf16(pa1, vf1[2], oa2);
        oa3 = wmma_bf16(pa0, vf0[3], oa3);
        oa3 = wmma_bf16(pa1, vf1[3], oa3);
        __syncthreads();
    }

    // normalize and write to [B, S, D] bf16 (D index = h*64 + d)
    f32x8 oas[4] = {oa0, oa1, oa2, oa3};
#pragma unroll
    for (int t = 0; t < 4; ++t) {
#pragma unroll
        for (int v = 0; v < 8; ++v) {
            int s = q0 + w * 16 + v + 8 * lh;
            int d = h * DK + t * 16 + lm;
            aob[((size_t)(b * SS + s)) * DD + d] = f32_to_bf16(oas[t][v] / lrun[v]);
        }
    }
}

// ---------------- launcher ----------------
extern "C" void kernel_launch(void* const* d_in, const int* in_sizes, int n_in,
                              void* d_out, int out_size, void* d_ws, size_t ws_size,
                              hipStream_t stream) {
    const float* x     = (const float*)d_in[0];
    const int*   tpos  = (const int*)d_in[1];
    const float* w_qkv = (const float*)d_in[2];
    const float* w_out = (const float*)d_in[3];
    float* out = (float*)d_out;

    const size_t nX = (size_t)BB * SS * DD;       // 4M
    const size_t nWQ = (size_t)3 * DD * DD;       // 3M
    const size_t nWO = (size_t)DD * DD;           // 1M

    char* ws = (char*)d_ws;
    unsigned short* xb    = (unsigned short*)(ws);
    unsigned short* wqkvb = (unsigned short*)(ws + 8388608);
    unsigned short* woutb = (unsigned short*)(ws + 14680064);
    unsigned short* qb    = (unsigned short*)(ws + 16777216);
    unsigned short* kb    = (unsigned short*)(ws + 25165824);
    unsigned short* vb    = (unsigned short*)(ws + 33554432);
    unsigned short* aob   = (unsigned short*)(ws + 41943040);

    // 1) convert inputs to bf16
    cvt_bf16_kernel<<<(nX + 255) / 256, 256, 0, stream>>>(x, xb, (int)nX);
    cvt_bf16_kernel<<<(nWQ + 255) / 256, 256, 0, stream>>>(w_qkv, wqkvb, (int)nWQ);
    cvt_bf16_kernel<<<(nWO + 255) / 256, 256, 0, stream>>>(w_out, woutb, (int)nWO);

    // 2) QKV projection: M=B*S=4096, N=3D=3072, K=D=1024; scatter into q/k/v heads
    {
        dim3 grid(3 * DD / 64, BB * SS / 64);
        gemm_bf16_kernel<<<grid, 128, 0, stream>>>(xb, wqkvb, 3 * DD, DD, 0,
                                                   qb, kb, vb, nullptr);
    }

    // 3) RoPE on q and k
    {
        int rows = BB * HH * SS;
        int threads = rows * 32;
        rope_kernel<<<(threads + 255) / 256, 256, 0, stream>>>(qb, tpos, rows);
        rope_kernel<<<(threads + 255) / 256, 256, 0, stream>>>(kb, tpos, rows);
    }

    // 4) causal flash attention
    {
        dim3 grid(SS / 64, HH, BB);
        fa_kernel<<<grid, 128, 0, stream>>>(qb, kb, vb, aob);
    }

    // 5) output projection: M=4096, N=D=1024, K=D=1024, fp32 out
    {
        dim3 grid(DD / 64, BB * SS / 64);
        gemm_bf16_kernel<<<grid, 128, 0, stream>>>(aob, woutb, DD, DD, 1,
                                                   nullptr, nullptr, nullptr, out);
    }
    (void)in_sizes; (void)n_in; (void)out_size; (void)ws_size;
}
